// MultiHeadAttention_45045617000604
// MI455X (gfx1250) — compile-verified, matrix operand order checked by reading, not run on hardware
//
#include <hip/hip_runtime.h>

// ---------------------------------------------------------------------------
// MI455X (gfx1250) fused multi-head attention, wave32 + v_wmma_f32_16x16x32_f16
// - 2x ds_load_b128 per f16 fragment (layout-matched LDS tiles)
// - 64x128 GEMM tiles, 8 WMMA per wave per staged 64-deep k-chunk
// - fragments preloaded in batches so ds_load latency overlaps WMMA issue
// - async global->LDS staging (ASYNCcnt) when the toolchain declares it
// ---------------------------------------------------------------------------

#define DEVI __device__ __forceinline__

typedef _Float16 v16h __attribute__((ext_vector_type(16)));
typedef _Float16 h8   __attribute__((ext_vector_type(8)));
typedef float    v8f  __attribute__((ext_vector_type(8)));
typedef int      i4v  __attribute__((vector_size(16)));   // builtin param type

constexpr int B  = 2;
constexpr int S  = 2048;
constexpr int D  = 1024;
constexpr int H  = 16;
constexpr int DH = 64;          // head dim (KS == VS == 64)
constexpr int BS = B * S;       // 4096 rows

constexpr int SSTRIDE = 2064;   // fp32 score-strip row stride (16B-aligned rows)
constexpr int KVSTR   = 72;     // f16 K chunk row stride  [128 keys][72]
constexpr int VTSTR   = 136;    // f16 V chunk row stride  [64 vd][136 keys]
constexpr size_t SMEM_ATTN =
    (size_t)16 * SSTRIDE * 4 +      // score/prob strip   (132096 B)
    (size_t)16 * DH * 2 +           // Q tile             (  2048 B)
    (size_t)128 * KVSTR * 2 +       // K/V chunk          ( 18432 B)
    (size_t)(128 + 32) * 4;         // reduction scratch  (   640 B)

#if defined(__HIP_DEVICE_COMPILE__) &&                                   \
    __has_builtin(__builtin_amdgcn_global_load_async_to_lds_b128) &&     \
    __has_builtin(__builtin_amdgcn_s_wait_asynccnt)
#define HAVE_ASYNC_LDS 1
#else
#define HAVE_ASYNC_LDS 0
#endif

#if HAVE_ASYNC_LDS
typedef __attribute__((address_space(1))) i4v* gptr_i4;
typedef __attribute__((address_space(3))) i4v* lptr_i4;
DEVI void async_copy16(void* lds_dst, const void* gsrc) {
  __builtin_amdgcn_global_load_async_to_lds_b128(
      (gptr_i4)gsrc, (lptr_i4)lds_dst, /*offset=*/0, /*cpol=*/0);
}
DEVI void async_wait() { __builtin_amdgcn_s_wait_asynccnt(0); }
#endif

DEVI v8f wmma_f16(v16h a, v16h b, v8f c) {
  // emits v_wmma_f32_16x16x32_f16
  return __builtin_amdgcn_wmma_f32_16x16x32_f16(
      /*neg_a=*/false, a, /*neg_b=*/false, b,
      /*c_mod=*/(short)0, c, /*reuse_a=*/false, /*reuse_b=*/false);
}

// 16-bit A/B fragment (ISA 7.12.2): per lane, elements are two contiguous
// 8 x f16 runs at K = kb..kb+7 and kb+16..kb+23, kb = k0 + 8*(lane>>4),
// major index (A: row M, B(n-major tile): col N) = maj0 + (lane&15).
// => two ds_load_b128 + one shuffle.
DEVI v16h frag16(const _Float16* s, int ld, int maj0, int k0) {
  const int lane = threadIdx.x & 31;
  const _Float16* p = s + (maj0 + (lane & 15)) * ld + k0 + ((lane >> 4) << 3);
  const h8 lo = *(const h8*)p;
  const h8 hi = *(const h8*)(p + 16);
  return __builtin_shufflevector(lo, hi, 0, 1, 2, 3, 4, 5, 6, 7,
                                 8, 9, 10, 11, 12, 13, 14, 15);
}

// Same fragment from an fp32 row-major tile (probabilities), cvt to f16.
DEVI v16h frag16_f32(const float* s, int ld, int maj0, int k0) {
  const int lane = threadIdx.x & 31;
  const float* p = s + (maj0 + (lane & 15)) * ld + k0 + ((lane >> 4) << 3);
  const float4 x0 = *(const float4*)p;
  const float4 x1 = *(const float4*)(p + 4);
  const float4 x2 = *(const float4*)(p + 16);
  const float4 x3 = *(const float4*)(p + 20);
  const h8 lo = {(_Float16)x0.x, (_Float16)x0.y, (_Float16)x0.z, (_Float16)x0.w,
                 (_Float16)x1.x, (_Float16)x1.y, (_Float16)x1.z, (_Float16)x1.w};
  const h8 hi = {(_Float16)x2.x, (_Float16)x2.y, (_Float16)x2.z, (_Float16)x2.w,
                 (_Float16)x3.x, (_Float16)x3.y, (_Float16)x3.z, (_Float16)x3.w};
  return __builtin_shufflevector(lo, hi, 0, 1, 2, 3, 4, 5, 6, 7,
                                 8, 9, 10, 11, 12, 13, 14, 15);
}

// ---------------------------------------------------------------------------
// Kernel 1: fused QKV projection.  X[4096,1024] x W[1024,1024] + bias.
// 64x128 tile, 64-deep k-chunks, 8 WMMA per wave per stage.
// Q,K -> f16 [B,H,S,64] ; V -> f16 TRANSPOSED [B,H,64,S] (feeds P@V B-frags).
// ---------------------------------------------------------------------------
__global__ __launch_bounds__(256) void qkv_proj_kernel(
    const float* __restrict__ q_in, const float* __restrict__ k_in,
    const float* __restrict__ v_in,
    const float* __restrict__ Wq, const float* __restrict__ Wk,
    const float* __restrict__ Wv,
    const float* __restrict__ bq, const float* __restrict__ bk,
    const float* __restrict__ bv,
    _Float16* __restrict__ Qh, _Float16* __restrict__ Kh,
    _Float16* __restrict__ Vt) {
  const int z = blockIdx.z;
  const float* X    = (z == 0) ? q_in : (z == 1) ? k_in : v_in;
  const float* W    = (z == 0) ? Wq : (z == 1) ? Wk : Wv;
  const float* bias = (z == 0) ? bq : (z == 1) ? bk : bv;
  _Float16* Out     = (z == 0) ? Qh : (z == 1) ? Kh : Vt;

  const int col0 = blockIdx.x * 128;
  const int row0 = blockIdx.y * 64;
  const int tid  = threadIdx.x;
  const int wid  = tid >> 5;
  const int lane = tid & 31;
  const int msub  = (wid & 3) * 16;   // 4 M-slots of 16
  const int nbase = (wid >> 2) * 64;  // 2 N-halves of 64

  __shared__ _Float16 sA[64 * 72];    // [m][k], stride 72, 64-deep k
  __shared__ _Float16 sWT[128 * 72];  // TRANSPOSED weights: [n][k], stride 72

  const v8f zero = {0.f, 0.f, 0.f, 0.f, 0.f, 0.f, 0.f, 0.f};
  v8f acc[4] = {zero, zero, zero, zero};

  for (int k0 = 0; k0 < D; k0 += 64) {
    __syncthreads();
#pragma unroll
    for (int j = 0; j < 2; ++j) {  // stage A (fp32 -> f16): 512 h8
      const int i = tid + j * 256;
      const int r = i >> 3, c8 = i & 7;
      const float4* src =
          (const float4*)&X[(size_t)(row0 + r) * D + k0 + c8 * 8];
      const float4 x0 = src[0], x1 = src[1];
      const h8 v = {(_Float16)x0.x, (_Float16)x0.y, (_Float16)x0.z,
                    (_Float16)x0.w, (_Float16)x1.x, (_Float16)x1.y,
                    (_Float16)x1.z, (_Float16)x1.w};
      *(h8*)&sA[r * 72 + c8 * 8] = v;
    }
#pragma unroll
    for (int j = 0; j < 8; ++j) {  // stage W transposed: [n][k], 2048 float4
      const int i = tid + j * 256;
      const int r = i >> 5, c4 = (i & 31) * 4;  // r = k row, c4 = n col group
      const float4 w = *(const float4*)&W[(size_t)(k0 + r) * D + col0 + c4];
      sWT[(c4 + 0) * 72 + r] = (_Float16)w.x;
      sWT[(c4 + 1) * 72 + r] = (_Float16)w.y;
      sWT[(c4 + 2) * 72 + r] = (_Float16)w.z;
      sWT[(c4 + 3) * 72 + r] = (_Float16)w.w;
    }
    if (k0 + 64 < D) {  // global_prefetch_b8
      __builtin_prefetch(&X[(size_t)(row0 + (tid >> 2)) * D + k0 + 64], 0, 1);
      __builtin_prefetch(&W[(size_t)(k0 + 64 + (tid >> 3)) * D + col0], 0, 1);
    }
    __syncthreads();
#pragma unroll
    for (int kh = 0; kh < 64; kh += 32) {   // A fragment reused across 4 B's
      const v16h a = frag16(sA, 72, msub, kh);
      v16h bf[4];
#pragma unroll
      for (int t = 0; t < 4; ++t)           // preload: all ds_loads in flight
        bf[t] = frag16(sWT, 72, nbase + t * 16, kh);
#pragma unroll
      for (int t = 0; t < 4; ++t)           // then back-to-back WMMA issue
        acc[t] = wmma_f16(a, bf[t], acc[t]);
    }
  }

  // C/D layout: lane&15 = N, VGPR r = M (+8 for upper lane half).
  const int n  = lane & 15;
  const int mo = (lane >> 4) * 8;
#pragma unroll
  for (int t = 0; t < 4; ++t) {
    const int coln = col0 + nbase + t * 16 + n;
    const float bb = bias[coln];
    const int hh = coln >> 6, dd = coln & 63;
#pragma unroll
    for (int r = 0; r < 8; ++r) {
      const int row = row0 + msub + mo + r;
      const int bidx = row >> 11, s = row & (S - 1);
      const _Float16 val = (_Float16)(acc[t][r] + bb);
      if (z == 2) {  // V: transposed per-head [B,H,64,S]
        Out[((size_t)(bidx * H + hh) * DH + dd) * S + s] = val;
      } else {       // Q,K: [B,H,S,64]
        Out[((size_t)(bidx * H + hh) * S + s) * DH + dd] = val;
      }
    }
  }
}

// ---------------------------------------------------------------------------
// Kernel 2: fused scores -> softmax -> attn write (NT) -> attn @ V.
// One WG = (b, h, 16-query strip), 128 threads = 4 waves.
// 16x2048 fp32 score strip stays in LDS; attn never re-read from HBM.
// ---------------------------------------------------------------------------
__global__ __launch_bounds__(128) void attention_kernel(
    const _Float16* __restrict__ Qh, const _Float16* __restrict__ Kh,
    const _Float16* __restrict__ Vt,
    float* __restrict__ attn, _Float16* __restrict__ ctx) {
  const int q0  = blockIdx.x * 16;
  const int h   = blockIdx.y;
  const int b   = blockIdx.z;
  const int tid = threadIdx.x;
  const int wid = tid >> 5;
  const int lane = tid & 31;

  extern __shared__ char smem_raw[];
  float*    sS   = (float*)smem_raw;                                  // 16 x SSTRIDE
  _Float16* sQ   = (_Float16*)(smem_raw + (size_t)16 * SSTRIDE * 4);  // 16 x 64
  _Float16* sKV  = sQ + 16 * DH;     // phase A: [128][KVSTR]; phase C: [64][VTSTR]
  float*    sRed = (float*)(sKV + 128 * KVSTR);                       // 128
  float*    sMax = sRed + 128;                                        // 16
  float*    sSum = sMax + 16;                                         // 16

  const size_t head = (size_t)(b * H + h) * S * DH;  // same count both layouts

  // ---- load Q tile (16x64 f16 = 128 x 16B) ----
#if HAVE_ASYNC_LDS
  async_copy16(&((h8*)sQ)[tid], &((const h8*)(Qh + head + (size_t)q0 * DH))[tid]);
  async_wait();
#else
  ((h8*)sQ)[tid] = ((const h8*)(Qh + head + (size_t)q0 * DH))[tid];
#endif
  __syncthreads();
  const v16h aQ0 = frag16(sQ, DH, 0, 0);
  const v16h aQ1 = frag16(sQ, DH, 0, 32);
  const v8f zero = {0.f, 0.f, 0.f, 0.f, 0.f, 0.f, 0.f, 0.f};

  // ---- Phase A: scores = (Q K^T) * 1/sqrt(64) ----
  for (int kc = 0; kc < S; kc += 128) {
    __syncthreads();
    {
      const h8* src = (const h8*)(Kh + head + (size_t)kc * DH);
      h8* dst = (h8*)sKV;
#pragma unroll 8
      for (int i = tid; i < 128 * 8; i += 128) {
#if HAVE_ASYNC_LDS
        async_copy16(&dst[(i >> 3) * (KVSTR / 8) + (i & 7)], &src[i]);
#else
        dst[(i >> 3) * (KVSTR / 8) + (i & 7)] = src[i];
#endif
      }
#if HAVE_ASYNC_LDS
      async_wait();
#endif
      if (kc + 128 < S)
        __builtin_prefetch(Kh + head + (size_t)(kc + 128) * DH + tid * 8, 0, 1);
    }
    __syncthreads();
#pragma unroll
    for (int t = 0; t < 2; ++t) {                 // each wave: 2 key tiles
      const int keyloc = (wid * 2 + t) * 16;
      const v16h b0 = frag16(sKV, KVSTR, keyloc, 0);   // K^T access, vectorized
      const v16h b1 = frag16(sKV, KVSTR, keyloc, 32);
      v8f c = zero;
      c = wmma_f16(aQ0, b0, c);
      c = wmma_f16(aQ1, b1, c);
      const int n = lane & 15, mo = (lane >> 4) * 8;
      const int key = kc + keyloc + n;
#pragma unroll
      for (int r = 0; r < 8; ++r)
        sS[(mo + r) * SSTRIDE + key] = c[r] * 0.125f;
    }
  }
  __syncthreads();

  // ---- softmax over each of 16 rows (2048 keys) ----
  {
    const int r = tid >> 3, g = tid & 7;
    float m = -3.4e38f;
    for (int k = g; k < S; k += 8) m = fmaxf(m, sS[r * SSTRIDE + k]);
    sRed[tid] = m;
    __syncthreads();
    if (g == 0) {
      float mm = sRed[tid];
#pragma unroll
      for (int j = 1; j < 8; ++j) mm = fmaxf(mm, sRed[tid + j]);
      sMax[r] = mm;
    }
    __syncthreads();
    const float rm = sMax[r];
    float sum = 0.f;
    for (int k = g; k < S; k += 8) {
      const float e = __expf(sS[r * SSTRIDE + k] - rm);
      sS[r * SSTRIDE + k] = e;
      sum += e;
    }
    sRed[tid] = sum;
    __syncthreads();
    if (g == 0) {
      float ss = 0.f;
#pragma unroll
      for (int j = 0; j < 8; ++j) ss += sRed[tid + j];
      sSum[r] = ss;
    }
    __syncthreads();
    // normalize in LDS + stream attn to HBM non-temporally (single-use data)
    const size_t attnBase = ((size_t)(b * H + h) * S + q0) * S;
    for (int i = tid; i < 16 * S; i += 128) {
      const int row = i >> 11, k = i & (S - 1);
      const float p = sS[row * SSTRIDE + k] / sSum[row];
      sS[row * SSTRIDE + k] = p;
      __builtin_nontemporal_store(p, &attn[attnBase + (size_t)row * S + k]);
    }
  }
  __syncthreads();

  // ---- Phase C: ctx = P @ V ; wave owns a 16-wide slice of head dim ----
  // V stored transposed [B,H,64,S] -> chunk stages as [64 vd][128 keys].
  v8f acc = zero;
  const int vd0 = wid * 16;
  for (int kc = 0; kc < S; kc += 128) {
    __syncthreads();
    {
      h8* dst = (h8*)sKV;  // reinterpreted [64][VTSTR]
#pragma unroll 8
      for (int i = tid; i < 64 * 16; i += 128) {
        const int row = i >> 4, c8 = i & 15;  // row = vd, 16 h8 per 128 keys
        const h8* src = &((const h8*)(Vt + head))[row * (S / 8) + (kc / 8) + c8];
#if HAVE_ASYNC_LDS
        async_copy16(&dst[row * (VTSTR / 8) + c8], src);
#else
        dst[row * (VTSTR / 8) + c8] = *src;
#endif
      }
#if HAVE_ASYNC_LDS
      async_wait();
#endif
      if (kc + 128 < S)
        __builtin_prefetch(Vt + head + (size_t)(tid >> 1) * S + kc + 128, 0, 1);
    }
    __syncthreads();
    // preload all 8 fragments for this chunk, then 4 back-to-back WMMAs
    v16h af[4], bf[4];
#pragma unroll
    for (int t = 0; t < 4; ++t) {
      af[t] = frag16_f32(sS, SSTRIDE, 0, kc + t * 32);  // fp32 probs -> f16
      bf[t] = frag16(sKV, VTSTR, vd0, t * 32);          // vectorized
    }
#pragma unroll
    for (int t = 0; t < 4; ++t) acc = wmma_f16(af[t], bf[t], acc);
  }
  // store ctx f16 into [B, S, H*64]
  {
    const int n = lane & 15, mo = (lane >> 4) * 8;
#pragma unroll
    for (int r = 0; r < 8; ++r) {
      const size_t dst =
          ((size_t)b * S + q0 + mo + r) * (size_t)(H * DH) + h * DH + vd0 + n;
      ctx[dst] = (_Float16)acc[r];
    }
  }
}

// ---------------------------------------------------------------------------
// Kernel 3: out = ctx(f16)[4096,1024] @ Wo[1024,1024] + bo  -> fp32
// 64x128 tile, 64-deep k-chunks, 8 WMMA per wave per stage.
// ---------------------------------------------------------------------------
__global__ __launch_bounds__(256) void out_proj_kernel(
    const _Float16* __restrict__ ctx, const float* __restrict__ Wo,
    const float* __restrict__ bo, float* __restrict__ out) {
  const int col0 = blockIdx.x * 128;
  const int row0 = blockIdx.y * 64;
  const int tid  = threadIdx.x;
  const int wid  = tid >> 5;
  const int lane = tid & 31;
  const int msub  = (wid & 3) * 16;
  const int nbase = (wid >> 2) * 64;

  __shared__ _Float16 sA[64 * 72];    // [m][k] stride 72
  __shared__ _Float16 sWT[128 * 72];  // transposed Wo: [n][k] stride 72

  const v8f zero = {0.f, 0.f, 0.f, 0.f, 0.f, 0.f, 0.f, 0.f};
  v8f acc[4] = {zero, zero, zero, zero};

  for (int k0 = 0; k0 < D; k0 += 64) {
    __syncthreads();
#pragma unroll
    for (int j = 0; j < 2; ++j) {  // stage ctx chunk: 512 h8
      const int i = tid + j * 256;
      const int r = i >> 3, c8 = i & 7;
      ((h8*)&sA[r * 72])[c8] =
          ((const h8*)ctx)[(size_t)(row0 + r) * (D / 8) + (k0 / 8) + c8];
    }
#pragma unroll
    for (int j = 0; j < 8; ++j) {  // stage Wo transposed: [n][k]
      const int i = tid + j * 256;
      const int r = i >> 5, c4 = (i & 31) * 4;
      const float4 w = *(const float4*)&Wo[(size_t)(k0 + r) * D + col0 + c4];
      sWT[(c4 + 0) * 72 + r] = (_Float16)w.x;
      sWT[(c4 + 1) * 72 + r] = (_Float16)w.y;
      sWT[(c4 + 2) * 72 + r] = (_Float16)w.z;
      sWT[(c4 + 3) * 72 + r] = (_Float16)w.w;
    }
    if (k0 + 64 < D)
      __builtin_prefetch(&Wo[(size_t)(k0 + 64 + (tid >> 3)) * D + col0], 0, 1);
    __syncthreads();
#pragma unroll
    for (int kh = 0; kh < 64; kh += 32) {
      const v16h a = frag16(sA, 72, msub, kh);
      v16h bf[4];
#pragma unroll
      for (int t = 0; t < 4; ++t)
        bf[t] = frag16(sWT, 72, nbase + t * 16, kh);
#pragma unroll
      for (int t = 0; t < 4; ++t)
        acc[t] = wmma_f16(a, bf[t], acc[t]);
    }
  }

  const int n  = lane & 15;
  const int mo = (lane >> 4) * 8;
#pragma unroll
  for (int t = 0; t < 4; ++t) {
    const int coln = col0 + nbase + t * 16 + n;
    const float bb = bo[coln];
#pragma unroll
    for (int r = 0; r < 8; ++r) {
      const int row = row0 + msub + mo + r;
      out[(size_t)row * D + coln] = acc[t][r] + bb;
    }
  }
}

// ---------------------------------------------------------------------------
extern "C" void kernel_launch(void* const* d_in, const int* in_sizes, int n_in,
                              void* d_out, int out_size, void* d_ws,
                              size_t ws_size, hipStream_t stream) {
  (void)in_sizes; (void)n_in; (void)out_size; (void)ws_size;

  const float* query = (const float*)d_in[0];
  const float* key_  = (const float*)d_in[1];
  const float* value = (const float*)d_in[2];
  const float* Wq = (const float*)d_in[3];
  const float* bq = (const float*)d_in[4];
  const float* Wk = (const float*)d_in[5];
  const float* bk = (const float*)d_in[6];
  const float* Wv = (const float*)d_in[7];
  const float* bv = (const float*)d_in[8];
  const float* Wo = (const float*)d_in[9];
  const float* bo = (const float*)d_in[10];

  float* out  = (float*)d_out;
  float* attn = out + (size_t)B * S * D;          // tuple: (out, attn)

  // workspace: Qh | Kh | Vt(transposed) | ctx, all f16 (32 MB total)
  _Float16* Qh  = (_Float16*)d_ws;
  _Float16* Kh  = Qh + (size_t)B * H * S * DH;
  _Float16* Vt  = Kh + (size_t)B * H * S * DH;
  _Float16* ctx = Vt + (size_t)B * H * S * DH;

  qkv_proj_kernel<<<dim3(D / 128, BS / 64, 3), 256, 0, stream>>>(
      query, key_, value, Wq, Wk, Wv, bq, bk, bv, Qh, Kh, Vt);

  attention_kernel<<<dim3(S / 16, H, B), 128, SMEM_ATTN, stream>>>(
      Qh, Kh, Vt, attn, ctx);

  out_proj_kernel<<<dim3(D / 128, BS / 64), 256, 0, stream>>>(ctx, Wo, bo, out);
}
